// SelfLinearAttentionCOSHead_37589553774714
// MI455X (gfx1250) — matrix operand order repaired and checked by reading, not compiled, not run on hardware
//
#include <hip/hip_runtime.h>
#include <hip/hip_bf16.h>
#include <math.h>

// Problem constants (from reference)
#define B_TOT    2048
#define S_LEN    71
#define SP       80        // S padded to 5 M-tiles of 16
#define SPK      96        // S padded to 3 K-steps of 32 (for kv GEMM)
#define D_MODEL  768
#define H_HEADS  4
#define DHD      192
#define TEXT_LEN 35
#define EPS_DEN  1e-6f

typedef __attribute__((ext_vector_type(16))) __bf16 v16bf;
typedef __attribute__((ext_vector_type(8)))  __bf16 v8bf;
typedef __attribute__((ext_vector_type(4)))  __bf16 v4bf;
typedef __attribute__((ext_vector_type(8)))  float  v8f;

// ---------------- workspace layout (bytes) ----------------
#define OFF_WQKV  0ull
#define SZ_WQKV   (2304ull * 768ull * 2ull)          // 3,538,944
#define OFF_WOUT  (OFF_WQKV + SZ_WQKV)
#define SZ_WOUT   (768ull * 768ull * 2ull)           // 1,179,648
#define OFF_MTB   (OFF_WOUT + SZ_WOUT)               // 4,718,592
#define SZ_MTB    (4096ull * 768ull * 2ull)          // 6,291,456 (rows 0..2047 = Mt, 2048.. = Mi)
#define OFF_R     (OFF_MTB + SZ_MTB)                 // 11,010,048
#define SZ_R      (4096ull * 768ull * 4ull)          // 12,582,912
#define OFF_SLAB  (OFF_R + SZ_R)                     // 23,592,960
#define Q_BYTES   30720ull                           // [80][192] bf16
#define KT_BYTES  36864ull                           // [192][96] bf16
#define HEAD_BYTES (Q_BYTES + 2ull*KT_BYTES)         // 104,448
#define SLOT_BYTES (HEAD_BYTES * 4ull)               // 417,792 per batch slot
#define CHUNK     512                                // slab: 512*417,792 ~ 214 MB

// LDS union offsets for the per-head phases of lin_attn
#define L_KT   0                       // 36,864  (also reused for qb: 30,720)
#define L_VT   36864                   // 36,864
#define L_KVT  73728                   // 73,728
#define L_Z    147456                  // 768
#define L_WGT  148224                  // 320
#define L_UNION_BYTES 148544           // >= 122,880 needed by phase-1 xs

__device__ __forceinline__ float phi_elu1(float x) {
  return x > 0.f ? x + 1.f : __expf(x);   // elu(x)+1
}

// A-fragment (16x32 bf16): lane<16 -> K{0..15}; lane>=16 -> K{16..31}
__device__ __forceinline__ v16bf load_a_frag(const __bf16* p) {
  v16bf a = *(const v16bf*)(p);
  return a;
}

// ---- CDNA5 async global->LDS DMA (ASYNCcnt-tracked), via inline asm ----
typedef __attribute__((address_space(3))) char lds_char;

__device__ __forceinline__ unsigned int lds_addr_of(void* p) {
  return (unsigned int)(unsigned long long)(lds_char*)p;
}

__device__ __forceinline__ void async_load_b128(unsigned int lds_byte_addr, const void* gaddr) {
  asm volatile("global_load_async_to_lds_b128 %0, %1, off"
               :: "v"(lds_byte_addr), "v"(gaddr)
               : "memory");
}

__device__ __forceinline__ void wait_async_all() {
  asm volatile("s_wait_asynccnt 0" ::: "memory");
}

// ---------------- kernel 0: f32 -> bf16 conversion of both weight matrices ----------------
__global__ void cvt_w_kernel(const float* __restrict__ wqkv, const float* __restrict__ wout,
                             __bf16* __restrict__ wqkv_b, __bf16* __restrict__ wout_b) {
  int i = blockIdx.x * 256 + threadIdx.x;
  const int nq = 2304 * 768, no = 768 * 768;
  if (i < nq) wqkv_b[i] = (__bf16)wqkv[i];
  else if (i < nq + no) wout_b[i - nq] = (__bf16)wout[i - nq];
}

// ---------------- kernel 1: fused linear attention per batch element ----------------
__global__ __launch_bounds__(256) void lin_attn_kernel(
    const float* __restrict__ x,       // [B,S,D] f32
    const float* __restrict__ b_qkv,   // [2304]  f32
    const __bf16* __restrict__ wq,     // [2304][768] bf16 (ws)
    char* __restrict__ slab_all,       // per-slot q/kT/vT scratch (ws)
    __bf16* __restrict__ Mtb,          // [4096][768] bf16 (ws): row b = Mt, row 2048+b = Mi
    int b_base)
{
  __shared__ __align__(32) char smemU[L_UNION_BYTES];
  __shared__ float mt[D_MODEL];
  __shared__ float mi[D_MODEL];

  __bf16* xs = (__bf16*)smemU;                 // phase 0/1: [80][768] bf16 = 122,880 B

  const int tid  = threadIdx.x;
  const int lane = tid & 31;
  const int wave = tid >> 5;
  const int l15  = lane & 15;
  const int aoff = (lane >> 4) * 16;
  const int boff = (lane >> 4) * 16;
  const int rofs = (lane >> 4) << 3;
  const int b    = b_base + blockIdx.x;

  char* slab = slab_all + (size_t)blockIdx.x * SLOT_BYTES;

  // -------- phase 0: stage x[b] as bf16 (vectorized); zero accumulators; zero slab K-pad --------
  for (int i = tid; i < D_MODEL; i += 256) { mt[i] = 0.f; mi[i] = 0.f; }
  const float* xb = x + (size_t)b * S_LEN * D_MODEL;
  {
    const int nv = SP * D_MODEL / 4;         // 15,360 groups of 4
    for (int i = tid; i < nv; i += 256) {
      int row = i / (D_MODEL / 4);
      int c4  = (i - row * (D_MODEL / 4)) * 4;
      v4bf o;
      if (row < S_LEN) {
        const float4 f = *(const float4*)(xb + (size_t)row * D_MODEL + c4);
        o[0] = (__bf16)f.x; o[1] = (__bf16)f.y; o[2] = (__bf16)f.z; o[3] = (__bf16)f.w;
      } else {
        o[0] = (__bf16)0.f; o[1] = (__bf16)0.f; o[2] = (__bf16)0.f; o[3] = (__bf16)0.f;
      }
      *(v4bf*)(xs + row * D_MODEL + c4) = o;
    }
  }
  for (int i = tid; i < H_HEADS * 2 * DHD * 16; i += 256) {   // kT/vT cols 80..95 = 0
    int h     = i / (2 * DHD * 16);
    int r     = i % (2 * DHD * 16);
    int which = r / (DHD * 16);
    int r2    = r % (DHD * 16);
    __bf16* arr = (__bf16*)(slab + (size_t)h * HEAD_BYTES + (which ? (Q_BYTES + KT_BYTES) : Q_BYTES));
    arr[(r2 >> 4) * SPK + 80 + (r2 & 15)] = (__bf16)0.f;
  }
  __syncthreads();

  // -------- phase 1: QKV = phi(x @ Wqkv^T + b), register-blocked over 5 M-tiles --------
  // Each wave owns N-strips; each B fragment is loaded once and used for 5 WMMAs.
  for (int strip = wave; strip < 144; strip += 8) {
    int e0 = strip * 16;
    int erow = e0 + l15;
    const __bf16* wrow = wq + (size_t)erow * D_MODEL;
    const __bf16* xcol = xs + l15 * D_MODEL;
    v8f acc[5];
#pragma unroll
    for (int m = 0; m < 5; ++m) acc[m] = (v8f){};
    for (int kk = 0; kk < 24; ++kk) {
      v16bf bf = *(const v16bf*)(wrow + kk * 32 + boff);
#pragma unroll
      for (int m = 0; m < 5; ++m) {
        v16bf a = load_a_frag(xcol + (m * 16) * D_MODEL + kk * 32 + aoff);
        acc[m] = __builtin_amdgcn_wmma_f32_16x16x32_bf16(false, a, false, bf, (short)0, acc[m], false, false);
      }
    }
    float bias = b_qkv[erow];
    int mtx = e0 / D_MODEL;             // 0=q, 1=k, 2=v
    int h   = (e0 % D_MODEL) / DHD;
    int col = (e0 % DHD) + l15;
    char* hb = slab + (size_t)h * HEAD_BYTES;
    __bf16* qb = (__bf16*)hb;
    __bf16* kT = (__bf16*)(hb + Q_BYTES);
    __bf16* vT = (__bf16*)(hb + Q_BYTES + KT_BYTES);
#pragma unroll
    for (int m = 0; m < 5; ++m) {
      int m0 = m * 16;
#pragma unroll
      for (int r = 0; r < 8; ++r) {
        int row = m0 + r + rofs;
        float val = acc[m][r] + bias;
        if (mtx == 0) {
          qb[row * DHD + col] = (__bf16)phi_elu1(val);
        } else if (mtx == 1) {
          kT[col * SPK + row] = (__bf16)((row < S_LEN) ? phi_elu1(val) : 0.f);
        } else {
          vT[col * SPK + row] = (__bf16)((row < S_LEN) ? val : 0.f);
        }
      }
    }
  }
  __threadfence_block();
  __syncthreads();

  // -------- per-head linear attention (all GEMM operands staged in LDS) --------
  __bf16* kTl  = (__bf16*)(smemU + L_KT);
  __bf16* vTl  = (__bf16*)(smemU + L_VT);
  __bf16* qbl  = (__bf16*)(smemU + L_KT);     // reuses kT/vT region after kv GEMM
  __bf16* kvTl = (__bf16*)(smemU + L_KVT);
  float*  zsh  = (float*)(smemU + L_Z);
  float*  wgt  = (float*)(smemU + L_WGT);

  for (int h = 0; h < H_HEADS; ++h) {
    char* hb = slab + (size_t)h * HEAD_BYTES;

    // stage kT+vT (contiguous 73,728 B) into LDS via async DMA
    {
      const char* src = hb + Q_BYTES;
      unsigned int dst = lds_addr_of(smemU + L_KT);
      for (int i = tid; i < (int)(2 * KT_BYTES / 16); i += 256)
        async_load_b128(dst + 16u * i, src + 16 * i);
      wait_async_all();
    }
    __syncthreads();

    // z[e] = sum_s k[s,e]
    if (tid < DHD) {
      float acc = 0.f;
      const __bf16* kr = kTl + tid * SPK;
#pragma unroll 8
      for (int s2 = 0; s2 < SPK; ++s2) acc += (float)kr[s2];
      zsh[tid] = acc;
    }
    __syncthreads();

    // kv GEMM: kvT[e2][e] = sum_s k[s,e] * v[s,e2]   (M=192,N=192,K=96)
    for (int t = wave; t < 144; t += 8) {
      int m0 = (t / 12) * 16, n0 = (t % 12) * 16;
      const __bf16* arow = kTl + (m0 + l15) * SPK;
      const __bf16* brow = vTl + (n0 + l15) * SPK;
      v8f c = {};
#pragma unroll
      for (int kk = 0; kk < 3; ++kk) {
        v16bf a  = load_a_frag(arow + kk * 32 + aoff);
        v16bf bf = *(const v16bf*)(brow + kk * 32 + boff);
        c = __builtin_amdgcn_wmma_f32_16x16x32_bf16(false, a, false, bf, (short)0, c, false, false);
      }
      __bf16* kvr = kvTl + (n0 + l15) * DHD;
#pragma unroll
      for (int r = 0; r < 8; ++r) kvr[m0 + r + rofs] = (__bf16)c[r];
    }
    __syncthreads();

    // stage qb (30,720 B) over the dead kT region via async DMA
    {
      unsigned int dst = lds_addr_of(smemU + L_KT);
      for (int i = tid; i < (int)(Q_BYTES / 16); i += 256)
        async_load_b128(dst + 16u * i, hb + 16 * i);
      wait_async_all();
    }
    __syncthreads();

    // wgt[s] = 1 / ((35|36) * (q[s].z + eps))
    if (tid < SP) {
      if (tid < S_LEN) {
        float acc = EPS_DEN;
        const __bf16* qr = qbl + tid * DHD;
#pragma unroll 8
        for (int e = 0; e < DHD; ++e) acc += (float)qr[e] * zsh[e];
        wgt[tid] = (tid < TEXT_LEN) ? 1.f / (35.f * acc) : 1.f / (36.f * acc);
      } else {
        wgt[tid] = 0.f;
      }
    }
    __syncthreads();

    // num GEMM + mean accumulation: num = q @ kv  (M=80,N=192,K=192)
    for (int t = wave; t < 60; t += 8) {
      int m0 = (t / 12) * 16, n0 = (t % 12) * 16;
      const __bf16* arow = qbl + (m0 + l15) * DHD;
      const __bf16* brow = kvTl + (n0 + l15) * DHD;
      v8f c = {};
#pragma unroll
      for (int kk = 0; kk < 6; ++kk) {
        v16bf a  = load_a_frag(arow + kk * 32 + aoff);
        v16bf bf = *(const v16bf*)(brow + kk * 32 + boff);
        c = __builtin_amdgcn_wmma_f32_16x16x32_bf16(false, a, false, bf, (short)0, c, false, false);
      }
      int col = h * DHD + n0 + l15;
#pragma unroll
      for (int r = 0; r < 8; ++r) {
        int s2 = m0 + r + rofs;
        if (s2 < S_LEN) {
          float v = c[r] * wgt[s2];
          atomicAdd((s2 < TEXT_LEN) ? &mt[col] : &mi[col], v);
        }
      }
    }
    __syncthreads();
  }

  for (int j = tid; j < D_MODEL; j += 256) {
    Mtb[(size_t)b * D_MODEL + j]          = (__bf16)mt[j];
    Mtb[(size_t)(b + 2048) * D_MODEL + j] = (__bf16)mi[j];
  }
}

// ---------------- kernel 2: R = tanh([Mt;Mi] @ w_out^T + b_out) via WMMA ----------------
// grid: (48 nTiles, 32 mGroups); each block: 8 waves x 1 M-tile each, shared w_out panel in LDS.
__global__ __launch_bounds__(256) void proj_tanh_kernel(
    const __bf16* __restrict__ Mtb,    // [4096][768] bf16
    const __bf16* __restrict__ woutb,  // [768][768] bf16
    const float* __restrict__ b_out,   // [768]
    float* __restrict__ R)             // [4096][768] f32
{
  __shared__ __align__(32) __bf16 wpanel[16 * D_MODEL];   // 24,576 B

  const int tid  = threadIdx.x;
  const int lane = tid & 31;
  const int wave = tid >> 5;
  const int l15  = lane & 15;
  const int aoff = (lane >> 4) * 16;
  const int boff = (lane >> 4) * 16;
  const int rofs = (lane >> 4) << 3;

  const int n0 = blockIdx.x * 16;
  const int m0 = (blockIdx.y * 8 + wave) * 16;

  // stage w_out rows [n0, n0+16) via async DMA
  {
    const char* src = (const char*)(woutb + (size_t)n0 * D_MODEL);
    unsigned int dst = lds_addr_of(wpanel);
    for (int i = tid; i < 16 * D_MODEL * 2 / 16; i += 256)
      async_load_b128(dst + 16u * i, src + 16 * i);
    wait_async_all();
  }
  __syncthreads();

  const __bf16* arow = Mtb + (size_t)(m0 + l15) * D_MODEL;
  const __bf16* brow = wpanel + l15 * D_MODEL;
  v8f c = {};
  for (int kk = 0; kk < 24; ++kk) {
    v16bf a  = load_a_frag(arow + kk * 32 + aoff);
    v16bf bf = *(const v16bf*)(brow + kk * 32 + boff);
    c = __builtin_amdgcn_wmma_f32_16x16x32_bf16(false, a, false, bf, (short)0, c, false, false);
  }
  int col = n0 + l15;
  float bias = b_out[col];
#pragma unroll
  for (int r = 0; r < 8; ++r) {
    int row = m0 + r + rofs;
    R[(size_t)row * D_MODEL + col] = tanhf(c[r] + bias);
  }
}

// ---------------- kernel 3: cosine similarity per batch element ----------------
__global__ __launch_bounds__(256) void cosine_kernel(const float* __restrict__ R,
                                                     float* __restrict__ out)
{
  __shared__ float red[3];
  int tid = threadIdx.x, b = blockIdx.x;
  if (tid < 3) red[tid] = 0.f;
  __syncthreads();
  const float* rt = R + (size_t)b * D_MODEL;
  const float* ri = R + (size_t)(b + 2048) * D_MODEL;
  float pd = 0.f, pt = 0.f, pi = 0.f;
  for (int j = tid; j < D_MODEL; j += 256) {
    float a = rt[j], c = ri[j];
    pd += a * c; pt += a * a; pi += c * c;
  }
  atomicAdd(&red[0], pd); atomicAdd(&red[1], pt); atomicAdd(&red[2], pi);
  __syncthreads();
  if (tid == 0) {
    float nt = fmaxf(sqrtf(red[1]), 1e-8f);
    float ni = fmaxf(sqrtf(red[2]), 1e-8f);
    out[b] = red[0] / (nt * ni);
  }
}

// ---------------- launcher ----------------
extern "C" void kernel_launch(void* const* d_in, const int* in_sizes, int n_in,
                              void* d_out, int out_size, void* d_ws, size_t ws_size,
                              hipStream_t stream) {
  const float* x      = (const float*)d_in[0];
  // d_in[1] = attention_mask (unused by the forward path)
  const float* w_qkv  = (const float*)d_in[2];
  const float* b_qkv  = (const float*)d_in[3];
  const float* w_out  = (const float*)d_in[4];
  const float* b_out  = (const float*)d_in[5];

  char*   ws    = (char*)d_ws;
  __bf16* wqb   = (__bf16*)(ws + OFF_WQKV);
  __bf16* woutb = (__bf16*)(ws + OFF_WOUT);
  __bf16* Mtb   = (__bf16*)(ws + OFF_MTB);
  float*  R     = (float*)(ws + OFF_R);
  char*   slab  = ws + OFF_SLAB;

  const int nCvt = 2304 * 768 + 768 * 768;
  cvt_w_kernel<<<(nCvt + 255) / 256, 256, 0, stream>>>(w_qkv, w_out, wqb, woutb);

  for (int c = 0; c < B_TOT / CHUNK; ++c) {
    lin_attn_kernel<<<CHUNK, 256, 0, stream>>>(x, b_qkv, wqb, slab, Mtb, c * CHUNK);
  }

  proj_tanh_kernel<<<dim3(48, 32), 256, 0, stream>>>(Mtb, woutb, b_out, R);
  cosine_kernel<<<B_TOT, 256, 0, stream>>>(R, (float*)d_out);
}
